// GatedMixer_80917183856720
// MI455X (gfx1250) — matrix operand order repaired and checked by reading, not compiled, not run on hardware
//
#include <hip/hip_runtime.h>
#include <hip/hip_bf16.h>

// ---------------------------------------------------------------------------
// Gated chunked attention for MI455X (gfx1250, wave32).
// All matmuls: v_wmma_f32_16x16x32_bf16 (bf16 operands, f32 accumulate).
// Fragment loads are 2x ds_load_b128 per operand (layouts chosen so every
// lane's fragment data is contiguous).  GEMM K-loop staged by the Tensor
// Data Mover (tensor_load_to_lds + s_wait_tensorcnt), double-buffered.
// ---------------------------------------------------------------------------

typedef __attribute__((ext_vector_type(16))) __bf16 v16bf;
typedef __attribute__((ext_vector_type(8)))  float  v8f;
typedef unsigned int u32x4 __attribute__((ext_vector_type(4)));
typedef int          i32x8 __attribute__((ext_vector_type(8)));
typedef int          i32x4 __attribute__((ext_vector_type(4)));

static constexpr int Bn  = 4;
static constexpr int Tn  = 4096;
static constexpr int Dn  = 1024;
static constexpr int Hn  = 8;
static constexpr int Cn  = 64;
static constexpr int NCn = Tn / Cn;      // 64 chunks
static constexpr int Mn  = Bn * Tn;      // 16384 rows
static constexpr int BHn = Bn * Hn;      // 32

#if defined(__gfx1250__) && __has_builtin(__builtin_amdgcn_tensor_load_to_lds) && \
    __has_builtin(__builtin_amdgcn_s_wait_tensorcnt)
#define USE_TDM 1
#else
#define USE_TDM 0
#endif

__device__ __forceinline__ v8f wmma_bf16(v16bf a, v16bf b, v8f c) {
  return __builtin_amdgcn_wmma_f32_16x16x32_bf16(false, a, false, b,
                                                 (short)0, c, false, false);
}

// A fragment 16x32 bf16 from row-major [m][k] (stride ld elems).
// lane half h=0/8: frag = row[h..h+7] ++ row[h+16..h+23]  (2x b128)
__device__ __forceinline__ v16bf fragA(const __bf16* p, int ld, int lane) {
  int m = lane & 15;
  int h = (lane >> 4) << 3;
  const __bf16* r = p + m * ld + h;
  union { v16bf v; uint4 q[2]; } u;
  u.q[0] = *(const uint4*)(r);
  u.q[1] = *(const uint4*)(r + 16);
  return u.v;
}

// B fragment 32x16 bf16 from n-major [n][k] (stride ld; B[k][n]=p[n*ld+k]).
// lane half h=0/8: frag = row[h..h+7] ++ row[h+16..h+23]  (2x b128)
__device__ __forceinline__ v16bf fragB_nm(const __bf16* p, int ld, int lane) {
  int n = lane & 15;
  int h = (lane >> 4) << 3;
  const __bf16* r = p + n * ld + h;
  union { v16bf v; uint4 q[2]; } u;
  u.q[0] = *(const uint4*)(r);
  u.q[1] = *(const uint4*)(r + 16);
  return u.v;
}

// C/D 16x16 f32: VGPR r -> row r / r+8 by lane half, col = lane&15.
__device__ __forceinline__ void store_tile(float* p, int ld, int lane, v8f a) {
  int n  = lane & 15;
  int ro = (lane >> 4) * 8;
#pragma unroll
  for (int r = 0; r < 8; ++r) p[(ro + r) * ld + n] = a[r];
}

// Pack 8 accumulator rows to 8 bf16 (16B) for contiguous transposed stores.
__device__ __forceinline__ uint4 pack8bf(v8f a) {
  union { uint4 q; __bf16 e[8]; } u;
#pragma unroll
  for (int r = 0; r < 8; ++r) u.e[r] = (__bf16)a[r];
  return u.q;
}

__device__ __forceinline__ v8f zero8() {
  v8f z = {0.f, 0.f, 0.f, 0.f, 0.f, 0.f, 0.f, 0.f};
  return z;
}

#if USE_TDM
// Issue one 2-D TDM tile load (bf16 elements): tile_d0 x tile_d1 with row
// stride stride0 (elems) from gaddr into LDS at lds_off; pads each 64B row
// with 16B so LDS row stride is 80B (= 40 bf16, matches frag alignment).
__device__ __forceinline__ void tdm_load_2d(const __bf16* gaddr, unsigned lds_off,
                                            unsigned tensor_d0, unsigned tensor_d1,
                                            unsigned tile_d0, unsigned tile_d1,
                                            unsigned stride0) {
  unsigned long long ga = (unsigned long long)(size_t)gaddr;
  u32x4 g0 = {1u,                                  // count=1, user mode
              lds_off,                             // LDS byte address
              (unsigned)ga,                        // global addr [31:0]
              (unsigned)((ga >> 32) & 0x1FFFFFFu) | (2u << 30)};  // addr[56:32] | type=2
  i32x8 g1 = {(int)((1u << 16) | (1u << 20) | (3u << 22) | (3u << 25)),
              //      data_size=2B  pad_en    pad_int=16DW  pad_amt=4DW
              (int)((tensor_d0 & 0xFFFFu) << 16),                 // tensor_dim0[15:0]
              (int)((tensor_d0 >> 16) | ((tensor_d1 & 0xFFFFu) << 16)),
              (int)((tensor_d1 >> 16) | (tile_d0 << 16)),
              (int)(tile_d1),                                     // tile_dim1 (tile_dim2=0)
              (int)(stride0),                                     // dim0 stride [31:0]
              0, 0};
  i32x4 z4 = {0, 0, 0, 0};
  i32x8 z8 = {0, 0, 0, 0, 0, 0, 0, 0};
  __builtin_amdgcn_tensor_load_to_lds(g0, g1, z4, z4, z8, 0);
}
#endif

// ---------------------------------------------------------------------------
__global__ void cvt_bf16_k(const float* __restrict__ s, __bf16* __restrict__ d,
                           size_t n) {
  size_t i      = (size_t)blockIdx.x * blockDim.x + threadIdx.x;
  size_t stride = (size_t)gridDim.x * blockDim.x;
  for (; i < n; i += stride) d[i] = (__bf16)s[i];
}

// ---------------------------------------------------------------------------
// Tiled WMMA GEMM:  C[M,N] = A[M,K] @ W[N,K]^T
// BM=BN=128, Kstep=32, 256 thr = 8 waves (each 32x64 = 2x4 tiles).
// MODE 0 (Q): RMS-norm * gain, bf16 row-major store (B,H,T,128)
// MODE 1 (K): RMS-norm, bf16 row-major + transposed (B,H,128,T) stores
// MODE 2 (V): plain, transposed store only
// MODE 3    : f32 row-major store (final output)
// ---------------------------------------------------------------------------
template <int MODE>
__global__ __launch_bounds__(256) void gemm_k(const __bf16* __restrict__ A,
                                              const __bf16* __restrict__ W,
                                              float* __restrict__ outF,
                                              __bf16* __restrict__ outB,
                                              __bf16* __restrict__ outBT,
                                              const float* __restrict__ gainPtr,
                                              int M, int N, int K) {
  constexpr int BM = 128, BN = 128, KS = 32, LD = 40;  // 32 + 8 pad elems
  __shared__ __align__(16) __bf16 As[2][BM * LD];
  __shared__ __align__(16) __bf16 Bs[2][BN * LD];
  __shared__ float rowsq[BM];

  int tid = threadIdx.x, lane = tid & 31, w = tid >> 5;
  int wr = w & 3, wc = w >> 2;
  int m0 = blockIdx.y * BM, n0 = blockIdx.x * BN;

  constexpr bool NORM = (MODE == 0 || MODE == 1);
  if (NORM && tid < BM) rowsq[tid] = 0.f;

  v8f acc[2][4];
#pragma unroll
  for (int mt = 0; mt < 2; ++mt)
#pragma unroll
    for (int nt = 0; nt < 4; ++nt) acc[mt][nt] = zero8();

  const int nk = K / KS;
#if USE_TDM
  if (w == 0) {
    tdm_load_2d(A + (size_t)m0 * K, (unsigned)(size_t)&As[0][0], K, M, KS, BM, K);
    tdm_load_2d(W + (size_t)n0 * K, (unsigned)(size_t)&Bs[0][0], K, N, KS, BN, K);
  }
#else
  int arow = tid >> 1;
  int ak   = (tid & 1) * 16;
  const __bf16* ag = A + (size_t)(m0 + arow) * K + ak;
  const __bf16* wg = W + (size_t)(n0 + arow) * K + ak;
#endif

  for (int i = 0; i < nk; ++i) {
    int buf = i & 1;
    __syncthreads();
#if USE_TDM
    if (w == 0) {
      if (i + 1 < nk) {
        int k1 = (i + 1) * KS;
        tdm_load_2d(A + (size_t)m0 * K + k1, (unsigned)(size_t)&As[buf ^ 1][0],
                    K, M, KS, BM, K);
        tdm_load_2d(W + (size_t)n0 * K + k1, (unsigned)(size_t)&Bs[buf ^ 1][0],
                    K, N, KS, BN, K);
        __builtin_amdgcn_s_wait_tensorcnt(2);   // stage i done, i+1 in flight
      } else {
        __builtin_amdgcn_s_wait_tensorcnt(0);
      }
    }
#else
    {
      int k0 = i * KS;
      *(uint4*)&As[buf][arow * LD + ak]     = *(const uint4*)(ag + k0);
      *(uint4*)&As[buf][arow * LD + ak + 8] = *(const uint4*)(ag + k0 + 8);
      *(uint4*)&Bs[buf][arow * LD + ak]     = *(const uint4*)(wg + k0);
      *(uint4*)&Bs[buf][arow * LD + ak + 8] = *(const uint4*)(wg + k0 + 8);
      if (k0 + KS < K) {
        __builtin_prefetch(ag + k0 + KS, 0, 0);
        __builtin_prefetch(wg + k0 + KS, 0, 0);
      }
    }
#endif
    __syncthreads();

    v16bf af[2], bfv[4];
#pragma unroll
    for (int mt = 0; mt < 2; ++mt)
      af[mt] = fragA(&As[buf][(wr * 32 + mt * 16) * LD], LD, lane);
#pragma unroll
    for (int nt = 0; nt < 4; ++nt)
      bfv[nt] = fragB_nm(&Bs[buf][(wc * 64 + nt * 16) * LD], LD, lane);
#pragma unroll
    for (int mt = 0; mt < 2; ++mt)
#pragma unroll
      for (int nt = 0; nt < 4; ++nt)
        acc[mt][nt] = wmma_bf16(af[mt], bfv[nt], acc[mt][nt]);
  }
  __syncthreads();

  int cn = lane & 15, ro = (lane >> 4) * 8;

  if (NORM) {
#pragma unroll
    for (int mt = 0; mt < 2; ++mt) {
      float s[8];
#pragma unroll
      for (int r = 0; r < 8; ++r) s[r] = 0.f;
#pragma unroll
      for (int nt = 0; nt < 4; ++nt)
#pragma unroll
        for (int r = 0; r < 8; ++r) {
          float v = acc[mt][nt][r];
          s[r] += v * v;
        }
#pragma unroll
      for (int r = 0; r < 8; ++r) {
        s[r] += __shfl_xor(s[r], 1, 32);
        s[r] += __shfl_xor(s[r], 2, 32);
        s[r] += __shfl_xor(s[r], 4, 32);
        s[r] += __shfl_xor(s[r], 8, 32);
      }
      if (cn == 0) {
#pragma unroll
        for (int r = 0; r < 8; ++r)
          atomicAdd(&rowsq[wr * 32 + mt * 16 + ro + r], s[r]);
      }
    }
    __syncthreads();
  }

  float gain = 1.f;
  if (MODE == 0 && gainPtr) gain = gainPtr[0];

#pragma unroll
  for (int mt = 0; mt < 2; ++mt) {
    int lrow0 = wr * 32 + mt * 16 + ro;
    int gm0 = m0 + lrow0;
    int b = gm0 / Tn, t0 = gm0 % Tn;
#pragma unroll
    for (int nt = 0; nt < 4; ++nt) {
      int gn = n0 + wc * 64 + nt * 16 + cn;
      int h = gn >> 7, d = gn & 127;
      v8f v;
#pragma unroll
      for (int r = 0; r < 8; ++r) {
        float x = acc[mt][nt][r];
        if (NORM)
          x *= rsqrtf(rowsq[lrow0 + r] * (1.f / 128.f) + 1e-6f) * gain;
        v[r] = x;
      }
      if (MODE == 3) {
#pragma unroll
        for (int r = 0; r < 8; ++r) outF[(size_t)(gm0 + r) * N + gn] = v[r];
      } else {
        if (MODE == 0 || MODE == 1) {  // row-major (B,H,T,128)
          size_t base = (((size_t)(b * Hn + h)) * Tn + t0) * 128 + d;
#pragma unroll
          for (int r = 0; r < 8; ++r) outB[base + (size_t)r * 128] = (__bf16)v[r];
        }
        if (MODE == 1 || MODE == 2) {  // transposed (B,H,128,T): 16B per lane
          *(uint4*)&outBT[(((size_t)(b * Hn + h)) * 128 + d) * Tn + t0] = pack8bf(v);
        }
      }
    }
  }
}

// ---------------------------------------------------------------------------
__global__ __launch_bounds__(256) void gate_k(const float* __restrict__ x,
                                              const float* __restrict__ Wg,
                                              const float* __restrict__ bg,
                                              float* __restrict__ g) {
  int m = blockIdx.x;
  int tid = threadIdx.x, lane = tid & 31, h = tid >> 5;
  const float* xr = x + (size_t)m * Dn;
  const float* wr = Wg + (size_t)h * Dn;
  float s = 0.f;
  for (int i = lane; i < Dn; i += 32) s += xr[i] * wr[i];
  for (int off = 16; off; off >>= 1) s += __shfl_xor(s, off, 32);
  if (lane == 0) g[(size_t)m * Hn + h] = 1.f / (1.f + __expf(-(s + bg[h])));
}

__global__ void gamma_k(const float* __restrict__ g, float* __restrict__ gm) {
  int idx = blockIdx.x * blockDim.x + threadIdx.x;
  if (idx >= BHn * NCn) return;
  int bh = idx / NCn, nc = idx % NCn;
  int b = bh / Hn, h = bh % Hn;
  float p = 1.f;
  for (int i = 0; i < Cn; ++i)
    p *= g[((size_t)(b * Tn + nc * Cn + i)) * Hn + h];
  gm[idx] = p;
}

// ---------------------------------------------------------------------------
// kv^T[e][d] = (k_chunk^T @ v_chunk)^T from transposed kT[d][t], vT[e][t].
// A = k^T row-major (kts), B = v n-major (vts): all frag loads b128.
// ---------------------------------------------------------------------------
__global__ __launch_bounds__(256) void kv_k(const __bf16* __restrict__ kT,
                                            const __bf16* __restrict__ vT,
                                            __bf16* __restrict__ kvb) {
  constexpr int LDT = 72;  // 64 + 8 pad
  __shared__ __align__(16) __bf16 kts[128 * LDT];
  __shared__ __align__(16) __bf16 vts[128 * LDT];
  int tid = threadIdx.x, lane = tid & 31, w = tid >> 5;
  int nc = blockIdx.x % NCn, bh = blockIdx.x / NCn;
  int row = tid >> 1, half = (tid & 1) * 32;
  size_t gbase = ((size_t)bh * 128 + row) * Tn + nc * Cn + half;
#pragma unroll
  for (int u = 0; u < 4; ++u) {
    *(uint4*)&kts[row * LDT + half + u * 8] = *(const uint4*)(kT + gbase + u * 8);
    *(uint4*)&vts[row * LDT + half + u * 8] = *(const uint4*)(vT + gbase + u * 8);
  }
  __syncthreads();

  v8f acc[8];
#pragma unroll
  for (int nt = 0; nt < 8; ++nt) acc[nt] = zero8();
#pragma unroll
  for (int kst = 0; kst < 2; ++kst) {
    v16bf af = fragA(&kts[(w * 16) * LDT + kst * 32], LDT, lane);
#pragma unroll
    for (int nt = 0; nt < 8; ++nt) {
      v16bf bv = fragB_nm(&vts[(nt * 16) * LDT + kst * 32], LDT, lane);
      acc[nt] = wmma_bf16(af, bv, acc[nt]);
    }
  }
  // transposed store [e][d]: 8 consecutive d per lane -> one b128 store
  __bf16* outp = kvb + (size_t)blockIdx.x * (128 * 128);
  int cn = lane & 15, ro = (lane >> 4) * 8;
#pragma unroll
  for (int nt = 0; nt < 8; ++nt)
    *(uint4*)&outp[(size_t)(nt * 16 + cn) * 128 + w * 16 + ro] = pack8bf(acc[nt]);
}

// ---------------------------------------------------------------------------
// Gated prefix scan over chunks per (b,h); state & kv both [e][d].
// ---------------------------------------------------------------------------
__global__ __launch_bounds__(256) void scan_k(const __bf16* __restrict__ kvb,
                                              const float* __restrict__ gm,
                                              __bf16* __restrict__ sb) {
  int bh = blockIdx.x, tid = threadIdx.x;
  float s[64];
#pragma unroll
  for (int j = 0; j < 64; ++j) s[j] = 0.f;
  for (int i = 0; i < NCn; ++i) {
    size_t base = ((size_t)bh * NCn + i) * (128 * 128);
#pragma unroll
    for (int j = 0; j < 64; ++j) sb[base + j * 256 + tid] = (__bf16)s[j];
    float g = gm[bh * NCn + i];
#pragma unroll
    for (int j = 0; j < 64; ++j)
      s[j] = s[j] * g + (float)kvb[base + j * 256 + tid];
  }
}

// ---------------------------------------------------------------------------
// Intra-chunk causal attention; v consumed transposed so probs@v B-frags
// are contiguous.  LDS regions overlapped (vT reuses q/k space).
// ---------------------------------------------------------------------------
__global__ __launch_bounds__(256) void attn_inner_k(const __bf16* __restrict__ qb,
                                                    const __bf16* __restrict__ kb,
                                                    const __bf16* __restrict__ vT,
                                                    float* __restrict__ yin) {
  constexpr int LD = 136, LDT = 72;
  __shared__ __align__(16) char smem[60416];
  __bf16* qs  = (__bf16*)smem;            // 17408 B
  __bf16* ks  = (__bf16*)(smem + 17408);  // 17408 B
  float*  sc  = (float*)(smem + 34816);   // 64x68 f32 = 17408 B
  __bf16* pb  = (__bf16*)(smem + 52224);  // 64x64 bf16 = 8192 B
  __bf16* vts = (__bf16*)smem;            // 128x72 = 18432 B (after scores)

  int tid = threadIdx.x, lane = tid & 31, w = tid >> 5;
  int nc = blockIdx.x % NCn, bh = blockIdx.x / NCn;
  size_t base = ((size_t)bh * Tn + nc * Cn) * 128;
  int tr = tid >> 2, seg = (tid & 3) * 32;
#pragma unroll
  for (int u = 0; u < 4; ++u) {
    *(uint4*)&qs[tr * LD + seg + u * 8] = *(const uint4*)(qb + base + tr * 128 + seg + u * 8);
    *(uint4*)&ks[tr * LD + seg + u * 8] = *(const uint4*)(kb + base + tr * 128 + seg + u * 8);
  }
  __syncthreads();

  // scores: 4x4 tiles; wave w -> tiles 2w, 2w+1
#pragma unroll
  for (int j = 0; j < 2; ++j) {
    int tile = w * 2 + j, mt = tile >> 2, nt = tile & 3;
    v8f a = zero8();
#pragma unroll
    for (int kst = 0; kst < 4; ++kst) {
      v16bf af = fragA(&qs[mt * 16 * LD + kst * 32], LD, lane);
      v16bf bv = fragB_nm(&ks[nt * 16 * LD + kst * 32], LD, lane);
      a = wmma_bf16(af, bv, a);
    }
    store_tile(&sc[(mt * 16) * 68 + nt * 16], 68, lane, a);
  }
  __syncthreads();

  // stage v^T (overlays q/k space) while 64 threads run the softmax
  int row = tid >> 1, half = (tid & 1) * 32;
  size_t vbase = ((size_t)bh * 128 + row) * Tn + nc * Cn + half;
#pragma unroll
  for (int u = 0; u < 4; ++u)
    *(uint4*)&vts[row * LDT + half + u * 8] = *(const uint4*)(vT + vbase + u * 8);

  if (tid < Cn) {
    int r = tid;
    const float qsc = 0.08838834764831845f;  // 1/sqrt(128)
    float mx = -3.4e38f;
    for (int s = 0; s <= r; ++s) {
      float v = sc[r * 68 + s] * qsc;
      sc[r * 68 + s] = v;
      mx = fmaxf(mx, v);
    }
    float sum = 0.f;
    for (int s = 0; s <= r; ++s) {
      float e = __expf(sc[r * 68 + s] - mx);
      sc[r * 68 + s] = e;
      sum += e;
    }
    float inv = 1.f / sum;
    for (int s = 0; s < Cn; ++s)
      pb[r * Cn + s] = (__bf16)((s <= r) ? sc[r * 68 + s] * inv : 0.f);
  }
  __syncthreads();

  // y_in = probs @ v : wave w -> mtile w&3, ntile group (w>>2)*4
  int mt = w & 3, ng = (w >> 2) * 4;
  v8f acc[4];
#pragma unroll
  for (int nt = 0; nt < 4; ++nt) acc[nt] = zero8();
#pragma unroll
  for (int kst = 0; kst < 2; ++kst) {
    v16bf af = fragA(&pb[mt * 16 * Cn + kst * 32], Cn, lane);
#pragma unroll
    for (int nt = 0; nt < 4; ++nt) {
      v16bf bv = fragB_nm(&vts[((ng + nt) * 16) * LDT + kst * 32], LDT, lane);
      acc[nt] = wmma_bf16(af, bv, acc[nt]);
    }
  }
  float* yo = yin + (size_t)blockIdx.x * Cn * 128;
#pragma unroll
  for (int nt = 0; nt < 4; ++nt)
    store_tile(&yo[(mt * 16) * 128 + (ng + nt) * 16], 128, lane, acc[nt]);
}

// ---------------------------------------------------------------------------
// y = y_in + q @ S_prefix; S stored transposed [e][d] so B-frags contiguous.
// ---------------------------------------------------------------------------
__global__ __launch_bounds__(256) void attn_cross_k(const __bf16* __restrict__ qb,
                                                    const __bf16* __restrict__ sb,
                                                    const float* __restrict__ yin,
                                                    __bf16* __restrict__ yb) {
  constexpr int LD = 136;
  __shared__ __align__(16) __bf16 qs[Cn * LD];
  __shared__ __align__(16) __bf16 Ss[128 * LD];
  int tid = threadIdx.x, lane = tid & 31, w = tid >> 5;
  int nc = blockIdx.x % NCn, bh = blockIdx.x / NCn;
  int b = bh / Hn, h = bh % Hn;
  size_t qbase = ((size_t)bh * Tn + nc * Cn) * 128;
  size_t sbase = (size_t)blockIdx.x * (128 * 128);
  int tr = tid >> 2, seg = (tid & 3) * 32;
#pragma unroll
  for (int u = 0; u < 4; ++u)
    *(uint4*)&qs[tr * LD + seg + u * 8] = *(const uint4*)(qb + qbase + tr * 128 + seg + u * 8);
  int sr = tid >> 1, ssg = (tid & 1) * 64;
#pragma unroll
  for (int u = 0; u < 8; ++u)
    *(uint4*)&Ss[sr * LD + ssg + u * 8] = *(const uint4*)(sb + sbase + sr * 128 + ssg + u * 8);
  __syncthreads();

  int mt = w & 3, ng = (w >> 2) * 4;
  v8f acc[4];
#pragma unroll
  for (int nt = 0; nt < 4; ++nt) acc[nt] = zero8();
#pragma unroll
  for (int kst = 0; kst < 4; ++kst) {
    v16bf af = fragA(&qs[mt * 16 * LD + kst * 32], LD, lane);
#pragma unroll
    for (int nt = 0; nt < 4; ++nt) {
      v16bf bv = fragB_nm(&Ss[((ng + nt) * 16) * LD + kst * 32], LD, lane);
      acc[nt] = wmma_bf16(af, bv, acc[nt]);
    }
  }
  const float* yi = yin + (size_t)blockIdx.x * Cn * 128;
  int cn = lane & 15, ro = (lane >> 4) * 8;
#pragma unroll
  for (int nt = 0; nt < 4; ++nt)
#pragma unroll
    for (int r = 0; r < 8; ++r) {
      int rowt = mt * 16 + ro + r;
      int col  = (ng + nt) * 16 + cn;
      float v = acc[nt][r] + yi[rowt * 128 + col];
      int t = nc * Cn + rowt;
      yb[((size_t)b * Tn + t) * (Hn * 128) + h * 128 + col] = (__bf16)v;
    }
}

// ---------------------------------------------------------------------------
extern "C" void kernel_launch(void* const* d_in, const int* in_sizes, int n_in,
                              void* d_out, int out_size, void* d_ws,
                              size_t ws_size, hipStream_t stream) {
  (void)in_sizes; (void)n_in; (void)out_size; (void)ws_size;
  const float* x  = (const float*)d_in[0];
  const float* Wq = (const float*)d_in[1];
  const float* Wk = (const float*)d_in[2];
  const float* Wv = (const float*)d_in[3];
  const float* Wg = (const float*)d_in[4];
  const float* bg = (const float*)d_in[5];
  const float* Wo = (const float*)d_in[6];
  const float* qg = (const float*)d_in[7];
  float* out = (float*)d_out;

  char* ws = (char*)d_ws;
  auto alloc = [&](size_t bytes) {
    char* p = ws;
    ws += (bytes + 255) & ~size_t(255);
    return p;
  };
  const size_t MK = (size_t)Mn * Dn;
  const size_t WW = (size_t)Dn * Dn;
  __bf16* xbf = (__bf16*)alloc(MK * 2);
  __bf16* wqb = (__bf16*)alloc(WW * 2);
  __bf16* wkb = (__bf16*)alloc(WW * 2);
  __bf16* wvb = (__bf16*)alloc(WW * 2);
  __bf16* wob = (__bf16*)alloc(WW * 2);
  __bf16* qbf = (__bf16*)alloc(MK * 2);   // (B,H,T,128)
  __bf16* kbf = (__bf16*)alloc(MK * 2);   // (B,H,T,128)
  __bf16* ktb = (__bf16*)alloc(MK * 2);   // (B,H,128,T)
  __bf16* vtb = (__bf16*)alloc(MK * 2);   // (B,H,128,T)
  float*  gbu = (float*)alloc((size_t)Mn * Hn * 4);
  float*  gmb = (float*)alloc((size_t)BHn * NCn * 4);
  __bf16* kvb = (__bf16*)alloc((size_t)BHn * NCn * 128 * 128 * 2);
  __bf16* sbu = (__bf16*)alloc((size_t)BHn * NCn * 128 * 128 * 2);
  float*  yin = (float*)alloc(MK * 4);
  __bf16* ybf = (__bf16*)alloc(MK * 2);

  cvt_bf16_k<<<8192, 256, 0, stream>>>(x, xbf, MK);
  cvt_bf16_k<<<1024, 256, 0, stream>>>(Wq, wqb, WW);
  cvt_bf16_k<<<1024, 256, 0, stream>>>(Wk, wkb, WW);
  cvt_bf16_k<<<1024, 256, 0, stream>>>(Wv, wvb, WW);
  cvt_bf16_k<<<1024, 256, 0, stream>>>(Wo, wob, WW);

  dim3 gg(Dn / 128, Mn / 128);
  gemm_k<0><<<gg, 256, 0, stream>>>(xbf, wqb, nullptr, qbf, nullptr, qg, Mn, Dn, Dn);
  gemm_k<1><<<gg, 256, 0, stream>>>(xbf, wkb, nullptr, kbf, ktb, nullptr, Mn, Dn, Dn);
  gemm_k<2><<<gg, 256, 0, stream>>>(xbf, wvb, nullptr, nullptr, vtb, nullptr, Mn, Dn, Dn);

  gate_k<<<Mn, 256, 0, stream>>>(x, Wg, bg, gbu);
  gamma_k<<<(BHn * NCn + 255) / 256, 256, 0, stream>>>(gbu, gmb);

  kv_k<<<BHn * NCn, 256, 0, stream>>>(ktb, vtb, kvb);
  scan_k<<<BHn, 256, 0, stream>>>(kvb, gmb, sbu);
  attn_inner_k<<<BHn * NCn, 256, 0, stream>>>(qbf, kbf, vtb, yin);
  attn_cross_k<<<BHn * NCn, 256, 0, stream>>>(qbf, sbu, yin, ybf);

  gemm_k<3><<<gg, 256, 0, stream>>>(ybf, wob, out, nullptr, nullptr, nullptr, Mn, Dn, Dn);
}
